// SpatialRefinementBlockV2_63067299775170
// MI455X (gfx1250) — compile-verified, matrix operand order checked by reading, not run on hardware
//
#include <hip/hip_runtime.h>
#include <hip/hip_bf16.h>
#include <math.h>
#include <stdint.h>

typedef _Float16 v16h __attribute__((ext_vector_type(16)));
typedef _Float16 v8h  __attribute__((ext_vector_type(8)));
typedef float    v8f  __attribute__((ext_vector_type(8)));

#define CC    256
#define BB    2
#define HI_H  64
#define HI_W  64
#define LO_H  128
#define LO_W  128
#define NPIX  (BB*LO_H*LO_W)    // 32768
#define NT32  (NPIX/32)         // 1024 tiles of 32 pixels

static inline int ceildiv(int a, int b) { return (a + b - 1) / b; }

// ---------------------------------------------------------------------------
// Small helper kernels
// ---------------------------------------------------------------------------

__global__ void zero_h16(_Float16* p, int n) {
    int i = blockIdx.x * blockDim.x + threadIdx.x;
    if (i < n) p[i] = (_Float16)0.f;
}

// OIHW f32 -> wT[tap][cout][cin] f16 (cout padded to COUTD)
__global__ void wprep(const float* __restrict__ w, _Float16* __restrict__ wT,
                      int COUTS, int CIN, int KH, int KW, int COUTD) {
    int i = blockIdx.x * blockDim.x + threadIdx.x;
    int total = COUTS * CIN * KH * KW;
    if (i >= total) return;
    int kx = i % KW; int t = i / KW;
    int ky = t % KH; t /= KH;
    int ci = t % CIN; int co = t / CIN;
    wT[((size_t)(ky * KW + kx) * COUTD + co) * CIN + ci] = (_Float16)w[i];
}

// NCHW f32 -> channels-last f16 with stride/offset
__global__ void nchw_to_cl(const float* __restrict__ in, _Float16* __restrict__ out,
                           int N, int Ch, int H, int W, int ostride, int choff) {
    int i = blockIdx.x * blockDim.x + threadIdx.x;
    int total = N * Ch * H * W;
    if (i >= total) return;
    int x = i % W; int t = i / W;
    int y = t % H; t /= H;
    int c = t % Ch; int b = t / Ch;
    out[((size_t)(b * H + y) * W + x) * ostride + choff + c] = (_Float16)in[i];
}

// depthwise 3x3 on 512-channel channels-last f16
__global__ void dw_conv(const _Float16* __restrict__ act, const float* __restrict__ w,
                        const float* __restrict__ bias, _Float16* __restrict__ out,
                        int H, int W, int npix) {
    int i = blockIdx.x * blockDim.x + threadIdx.x;
    if (i >= npix * 512) return;
    int c = i & 511; int pix = i >> 9;
    int hw = H * W; int b = pix / hw; int rem = pix - b * hw;
    int y = rem / W; int x = rem - y * W;
    float acc = bias[c];
    const float* wc = w + c * 9;
    for (int ky = 0; ky < 3; ++ky) {
        int yy = y + ky - 1; if (yy < 0 || yy >= H) continue;
        for (int kx = 0; kx < 3; ++kx) {
            int xx = x + kx - 1; if (xx < 0 || xx >= W) continue;
            acc += (float)act[((size_t)(b * H + yy) * W + xx) * 512 + c] * wc[ky * 3 + kx];
        }
    }
    out[(size_t)pix * 512 + c] = (_Float16)acc;
}

// 3x3 conv 768 -> 1, + sigmoid  (tiny: 226 MMAC)
__global__ void wt_conv(const _Float16* __restrict__ dir, const float* __restrict__ w,
                        const float* __restrict__ bias, float* __restrict__ wgt,
                        int H, int W, int npix) {
    int pix = blockIdx.x * blockDim.x + threadIdx.x;
    if (pix >= npix) return;
    int hw = H * W; int b = pix / hw; int rem = pix - b * hw;
    int y = rem / W; int x = rem - y * W;
    float acc = bias[0];
    for (int ky = 0; ky < 3; ++ky) {
        int yy = y + ky - 1; if (yy < 0 || yy >= H) continue;
        for (int kx = 0; kx < 3; ++kx) {
            int xx = x + kx - 1; if (xx < 0 || xx >= W) continue;
            const _Float16* p = dir + ((size_t)(b * H + yy) * W + xx) * 768;
            const float* wk = w + ky * 3 + kx;   // w[(c*9) + tap]
            float s = 0.f;
            for (int c = 0; c < 768; ++c) s += (float)p[c] * wk[c * 9];
            acc += s;
        }
    }
    wgt[pix] = 1.f / (1.f + expf(-acc));
}

// ---------------------------------------------------------------------------
// Generic implicit-GEMM conv via WMMA.
// Tile: 32 pixels x 64 out-channels, K slab of 64 per barrier round.
// A staged by all 128 threads (2x 16B each); B staged via async DMA to LDS
// (global_load_async_to_lds_b128, ASYNCcnt). 4 WMMAs per wave per round.
// ---------------------------------------------------------------------------
__global__ __launch_bounds__(128)
void conv_gemm(const _Float16* __restrict__ act, int act_stride, int act_choff,
               int in_H, int in_W, int deconv_mode,
               const _Float16* __restrict__ wT, const float* __restrict__ bias, int bias_n,
               int CIN, int COUT, int KH, int KW, int dil, int pad_h, int pad_w,
               int H, int W,
               int out_mode, _Float16* __restrict__ out_h, int out_stride, int out_choff,
               float* __restrict__ out_f) {
    __shared__ __align__(16) _Float16 ldsA[32][72];
    __shared__ __align__(16) _Float16 ldsB[64][72];
    const int tid  = threadIdx.x;
    const int lane = tid & 31;
    const int wave = tid >> 5;
    const int tile_pix = blockIdx.x * 32;
    const int HW = H * W;
    const int coutBlk = blockIdx.y * 64;

    const int coutn  = coutBlk + wave * 16 + (lane & 15);
    const bool active = (coutBlk + wave * 16) < COUT;

    v8f acc0 = {}; v8f acc1 = {};

    // A staging: 2 units/thread: u = uu*128 + tid; row = u>>3 (0..31), seg = u&7
    int srow[2], sseg[2], sy[2], sx[2], sb[2];
#pragma unroll
    for (int uu = 0; uu < 2; ++uu) {
        int u = uu * 128 + tid;
        srow[uu] = u >> 3; sseg[uu] = u & 7;
        int p = tile_pix + srow[uu];
        sb[uu] = p / HW; int rem = p - sb[uu] * HW;
        sy[uu] = rem / W; sx[uu] = rem - sy[uu] * W;
    }
    // B staging: 4 units/thread: row = u>>3 (0..63), seg = u&7
    int brow[4], bseg[4];
    uint32_t boff[4];
#pragma unroll
    for (int uu = 0; uu < 4; ++uu) {
        int u = uu * 128 + tid;
        brow[uu] = u >> 3; bseg[uu] = u & 7;
        boff[uu] = (uint32_t)(size_t)&ldsB[brow[uu]][bseg[uu] * 8];
    }

    const int arow  = lane & 15;
    const int abase = (lane >> 4) * 8;
    const int bR    = wave * 16 + (lane & 15);
    const int KT  = KH * KW;
    const int NC2 = CIN >> 6;   // 64-wide K rounds

    for (int tap = 0; tap < KT; ++tap) {
        const int dy = (tap / KW) * dil - pad_h;
        const int dx = (tap % KW) * dil - pad_w;
        const size_t wrowbase = (size_t)(tap * COUT);
        for (int ck = 0; ck < NC2; ++ck) {
            const int cin0 = ck * 64;
            // ---- async B tile: 64 cout rows x 64 cin, weights always valid ----
#pragma unroll
            for (int uu = 0; uu < 4; ++uu) {
                int rc = coutBlk + brow[uu];
                rc = (rc < COUT) ? rc : (COUT - 1);
                const _Float16* gp = wT + (wrowbase + rc) * CIN + cin0 + bseg[uu] * 8;
                unsigned long long ga = (unsigned long long)(size_t)gp;
                asm volatile("global_load_async_to_lds_b128 %0, %1, off"
                             :: "v"(boff[uu]), "v"(ga) : "memory");
            }
            // ---- A tile: 32 pixel rows x 64 cin, zero-fill at borders ----
#pragma unroll
            for (int uu = 0; uu < 2; ++uu) {
                v8h val = {};
                int yy = sy[uu] + dy, xx = sx[uu] + dx;
                bool v;
                if (deconv_mode) {
                    v = (yy >= 0) && (xx >= 0) && !(yy & 1) && !(xx & 1);
                    yy >>= 1; xx >>= 1;
                    v = v && (yy < in_H) && (xx < in_W);
                } else {
                    v = (yy >= 0) && (yy < in_H) && (xx >= 0) && (xx < in_W);
                }
                if (v) {
                    const size_t src = (size_t)(sb[uu] * in_H + yy) * in_W + xx;
                    val = *(const v8h*)(act + src * act_stride + act_choff + cin0 + sseg[uu] * 8);
                }
                *(v8h*)&ldsA[srow[uu]][sseg[uu] * 8] = val;
            }
            asm volatile("s_wait_asynccnt 0" ::: "memory");
            __syncthreads();
            if (active) {
#pragma unroll
                for (int kk = 0; kk < 2; ++kk) {
                    const int kb = kk * 32 + abase;
                    union { v16h v; v8h h[2]; } A0, A1, Bv;
                    A0.h[0] = *(const v8h*)&ldsA[arow][kb];
                    A0.h[1] = *(const v8h*)&ldsA[arow][kb + 16];
                    A1.h[0] = *(const v8h*)&ldsA[arow + 16][kb];
                    A1.h[1] = *(const v8h*)&ldsA[arow + 16][kb + 16];
                    Bv.h[0] = *(const v8h*)&ldsB[bR][kb];
                    Bv.h[1] = *(const v8h*)&ldsB[bR][kb + 16];
                    acc0 = __builtin_amdgcn_wmma_f32_16x16x32_f16(
                        false, A0.v, false, Bv.v, (short)0, acc0, false, false);
                    acc1 = __builtin_amdgcn_wmma_f32_16x16x32_f16(
                        false, A1.v, false, Bv.v, (short)0, acc1, false, false);
                }
            }
            __syncthreads();
        }
    }

    if (active && coutn < COUT) {
        const float bv = (bias && coutn < bias_n) ? bias[coutn] : 0.f;
        const int mbase = (lane >> 4) * 8;   // C/D layout: vgpr r, lanes>=16 -> M=8+r
#pragma unroll
        for (int r = 0; r < 8; ++r) {
#pragma unroll
            for (int s = 0; s < 2; ++s) {
                const int pix = tile_pix + s * 16 + mbase + r;
                const float val = (s ? acc1[r] : acc0[r]) + bv;
                if (out_mode == 0) {
                    out_h[(size_t)pix * out_stride + out_choff + coutn] = (_Float16)val;
                } else if (out_mode == 1) {
                    out_f[(size_t)pix * out_stride + coutn] = val;
                } else { // NCHW f32
                    const int b = pix / HW; const int rem = pix - b * HW;
                    out_f[(size_t)(b * COUT + coutn) * HW + rem] = val;
                }
            }
        }
    }
}

// ---------------------------------------------------------------------------
// Deformable conv (DCNv1) via WMMA: bilinear-sample fh_up into an LDS A tile
// (32 px x 256 cin) per tap, then 16 WMMAs per wave per tap (B shared across
// the two M-subtiles). Fused epilogue: mod = fh_sampled*sigmoid_w + f_low.
// ---------------------------------------------------------------------------
__global__ __launch_bounds__(128)
void deform_gemm(const _Float16* __restrict__ cat,   // stride 512: [0:256)=fh_up, [256:512)=f_low
                 const float* __restrict__ offs,      // [pix*32 + 2k(+0 dy,+1 dx)]
                 const float* __restrict__ wgt,       // [pix]
                 const _Float16* __restrict__ wT,     // [9][256][256]
                 _Float16* __restrict__ modbuf,       // [pix*256]
                 int H, int W) {
    __shared__ __align__(16) _Float16 ldsA[32][256];
    const int tid  = threadIdx.x;
    const int lane = tid & 31;
    const int wave = tid >> 5;
    const int tile_pix = blockIdx.x * 32;
    const int HW = H * W;

    const int coutn = blockIdx.y * 64 + wave * 16 + (lane & 15);

    v8f acc0 = {}; v8f acc1 = {};

    // staging: 4 threads per row, 64 channels each
    const int srow   = tid >> 2;
    const int schunk = tid & 3;
    const int sp = tile_pix + srow;
    const int sb = sp / HW;
    const int srem = sp - sb * HW;
    const int sy = srem / W;
    const int sx = srem - sy * W;

    const int arow  = lane & 15;
    const int abase = (lane >> 4) * 8;

    for (int k = 0; k < 9; ++k) {
        {   // bilinear sample this tap into LDS
            const float dyo = offs[(size_t)sp * 32 + k * 2 + 0];
            const float dxo = offs[(size_t)sp * 32 + k * 2 + 1];
            const float py = (float)sy + (float)(k / 3 - 1) + dyo;
            const float px = (float)sx + (float)(k % 3 - 1) + dxo;
            const float y0f = floorf(py), x0f = floorf(px);
            const float ly = py - y0f, lx = px - x0f;
            const int y0 = (int)y0f, x0 = (int)x0f;
            float w00 = (1.f - ly) * (1.f - lx), w01 = (1.f - ly) * lx;
            float w10 = ly * (1.f - lx),         w11 = ly * lx;
            const int y0c = min(max(y0, 0), H - 1), y1c = min(max(y0 + 1, 0), H - 1);
            const int x0c = min(max(x0, 0), W - 1), x1c = min(max(x0 + 1, 0), W - 1);
            if (y0 < 0 || y0 >= H)         { w00 = 0.f; w01 = 0.f; }
            if (y0 + 1 < 0 || y0 + 1 >= H) { w10 = 0.f; w11 = 0.f; }
            if (x0 < 0 || x0 >= W)         { w00 = 0.f; w10 = 0.f; }
            if (x0 + 1 < 0 || x0 + 1 >= W) { w01 = 0.f; w11 = 0.f; }
            const _Float16* p00 = cat + ((size_t)(sb * H + y0c) * W + x0c) * 512;
            const _Float16* p01 = cat + ((size_t)(sb * H + y0c) * W + x1c) * 512;
            const _Float16* p10 = cat + ((size_t)(sb * H + y1c) * W + x0c) * 512;
            const _Float16* p11 = cat + ((size_t)(sb * H + y1c) * W + x1c) * 512;
            const int cb = schunk * 64;
            for (int c = 0; c < 64; ++c) {
                float v = w00 * (float)p00[cb + c] + w01 * (float)p01[cb + c]
                        + w10 * (float)p10[cb + c] + w11 * (float)p11[cb + c];
                ldsA[srow][cb + c] = (_Float16)v;
            }
        }
        __syncthreads();
        const _Float16* wtap = wT + ((size_t)k * 256 + coutn) * 256;
#pragma unroll
        for (int ck = 0; ck < 8; ++ck) {
            union { v16h v; v8h h[2]; } A0, A1, Bv;
            const int kb = ck * 32 + abase;
            const _Float16* bp = wtap + ck * 32 + abase;
            Bv.h[0] = *(const v8h*)bp;
            Bv.h[1] = *(const v8h*)(bp + 16);
            __builtin_prefetch(bp + 32, 0, 3);
            A0.h[0] = *(const v8h*)&ldsA[arow][kb];
            A0.h[1] = *(const v8h*)&ldsA[arow][kb + 16];
            A1.h[0] = *(const v8h*)&ldsA[arow + 16][kb];
            A1.h[1] = *(const v8h*)&ldsA[arow + 16][kb + 16];
            acc0 = __builtin_amdgcn_wmma_f32_16x16x32_f16(
                false, A0.v, false, Bv.v, (short)0, acc0, false, false);
            acc1 = __builtin_amdgcn_wmma_f32_16x16x32_f16(
                false, A1.v, false, Bv.v, (short)0, acc1, false, false);
        }
        __syncthreads();
    }

    const int mbase = (lane >> 4) * 8;
#pragma unroll
    for (int r = 0; r < 8; ++r) {
#pragma unroll
        for (int s = 0; s < 2; ++s) {
            const int pix = tile_pix + s * 16 + mbase + r;
            const float a = s ? acc1[r] : acc0[r];
            const float v = a * wgt[pix] + (float)cat[(size_t)pix * 512 + 256 + coutn];
            modbuf[(size_t)pix * 256 + coutn] = (_Float16)v;
        }
    }
}

// ---------------------------------------------------------------------------
// Host launcher
// ---------------------------------------------------------------------------
extern "C" void kernel_launch(void* const* d_in, const int* in_sizes, int n_in,
                              void* d_out, int out_size, void* d_ws, size_t ws_size,
                              hipStream_t stream) {
    (void)in_sizes; (void)n_in; (void)out_size; (void)ws_size;

    const float* f_high   = (const float*)d_in[0];
    const float* f_low    = (const float*)d_in[1];
    const float* w_deconv = (const float*)d_in[2];
    const float* b_deconv = (const float*)d_in[3];
    const float* w_offset = (const float*)d_in[4];
    const float* b_offset = (const float*)d_in[5];
    const float* w_dw     = (const float*)d_in[6];
    const float* b_dw     = (const float*)d_in[7];
    const float* w_pw     = (const float*)d_in[8];
    const float* b_pw     = (const float*)d_in[9];
    const float* w_dir1   = (const float*)d_in[10];
    const float* b_dir1   = (const float*)d_in[11];
    const float* w_dir2   = (const float*)d_in[12];
    const float* b_dir2   = (const float*)d_in[13];
    const float* w_dir3   = (const float*)d_in[14];
    const float* b_dir3   = (const float*)d_in[15];
    const float* w_wt     = (const float*)d_in[16];
    const float* b_wt     = (const float*)d_in[17];
    const float* w_deform = (const float*)d_in[18];
    const float* w_final  = (const float*)d_in[19];
    const float* b_final  = (const float*)d_in[20];
    float* out = (float*)d_out;

    // workspace layout (all sizes multiples of 256B)
    char* ws = (char*)d_ws;
    const size_t SZ_CAT  = (size_t)NPIX * 512 * 2;
    const size_t SZ_DIR  = (size_t)NPIX * 768 * 2;
    const size_t SZ_FEAT = (size_t)NPIX * 256 * 2;
    const size_t SZ_DW   = (size_t)NPIX * 512 * 2;
    const size_t SZ_OFFS = (size_t)NPIX * 32 * 4;   // == fhi size, aliased
    const size_t SZ_WGT  = (size_t)NPIX * 4;

    size_t o = 0;
    _Float16* cat    = (_Float16*)(ws + o); o += SZ_CAT;
    _Float16* dir    = (_Float16*)(ws + o); o += SZ_DIR;
    _Float16* feat1  = (_Float16*)(ws + o); o += SZ_FEAT;
    _Float16* dwbuf  = (_Float16*)(ws + o);
    _Float16* modbuf = dwbuf;               o += SZ_DW;     // modbuf aliases dwbuf (dead after pw)
    _Float16* fhi    = (_Float16*)(ws + o);
    float*    offsb  = (float*)(ws + o);    o += SZ_OFFS;   // offs aliases fhi (dead after deconv)
    float*    wgtb   = (float*)(ws + o);    o += SZ_WGT;
    _Float16* wT_deconv = (_Float16*)(ws + o); o += (size_t)9 * 256 * 256 * 2;
    _Float16* wT_offset = (_Float16*)(ws + o); o += (size_t)9 * 32 * 512 * 2;
    _Float16* wT_pw     = (_Float16*)(ws + o); o += (size_t)1 * 256 * 512 * 2;
    _Float16* wT_d1     = (_Float16*)(ws + o); o += (size_t)3 * 256 * 256 * 2;
    _Float16* wT_d2     = (_Float16*)(ws + o); o += (size_t)3 * 256 * 256 * 2;
    _Float16* wT_d3     = (_Float16*)(ws + o); o += (size_t)9 * 256 * 256 * 2;
    _Float16* wT_deform = (_Float16*)(ws + o); o += (size_t)9 * 256 * 256 * 2;
    _Float16* wT_final  = (_Float16*)(ws + o); o += (size_t)9 * 256 * 256 * 2;

    // 1) weight prep
    wprep<<<ceildiv(256*256*9, 256), 256, 0, stream>>>(w_deconv, wT_deconv, 256, 256, 3, 3, 256);
    zero_h16<<<ceildiv(9*32*512, 256), 256, 0, stream>>>(wT_offset, 9*32*512);
    wprep<<<ceildiv(18*512*9, 256), 256, 0, stream>>>(w_offset, wT_offset, 18, 512, 3, 3, 32);
    wprep<<<ceildiv(256*512, 256), 256, 0, stream>>>(w_pw, wT_pw, 256, 512, 1, 1, 256);
    wprep<<<ceildiv(256*256*3, 256), 256, 0, stream>>>(w_dir1, wT_d1, 256, 256, 1, 3, 256);
    wprep<<<ceildiv(256*256*3, 256), 256, 0, stream>>>(w_dir2, wT_d2, 256, 256, 3, 1, 256);
    wprep<<<ceildiv(256*256*9, 256), 256, 0, stream>>>(w_dir3, wT_d3, 256, 256, 3, 3, 256);
    wprep<<<ceildiv(256*256*9, 256), 256, 0, stream>>>(w_deform, wT_deform, 256, 256, 3, 3, 256);
    wprep<<<ceildiv(256*256*9, 256), 256, 0, stream>>>(w_final, wT_final, 256, 256, 3, 3, 256);

    // 2) activation layout conversion
    nchw_to_cl<<<ceildiv(BB*256*HI_H*HI_W, 256), 256, 0, stream>>>(
        f_high, fhi, BB, 256, HI_H, HI_W, 256, 0);
    nchw_to_cl<<<ceildiv(BB*256*LO_H*LO_W, 256), 256, 0, stream>>>(
        f_low, cat, BB, 256, LO_H, LO_W, 512, 256);

    // 3) deconv (stride-2 upsample) -> cat[:, 0:256)
    conv_gemm<<<dim3(NT32, 4), 128, 0, stream>>>(
        fhi, 256, 0, HI_H, HI_W, 1, wT_deconv, b_deconv, 256,
        256, 256, 3, 3, 1, 1, 1, LO_H, LO_W, 0, cat, 512, 0, nullptr);

    // 4) offset conv 512 -> 18 (padded 32), f32 out
    conv_gemm<<<dim3(NT32, 1), 128, 0, stream>>>(
        cat, 512, 0, LO_H, LO_W, 0, wT_offset, b_offset, 18,
        512, 32, 3, 3, 1, 1, 1, LO_H, LO_W, 1, nullptr, 32, 0, offsb);

    // 5) depthwise 3x3
    dw_conv<<<ceildiv(NPIX*512, 256), 256, 0, stream>>>(cat, w_dw, b_dw, dwbuf, LO_H, LO_W, NPIX);

    // 6) pointwise 512 -> 256
    conv_gemm<<<dim3(NT32, 4), 128, 0, stream>>>(
        dwbuf, 512, 0, LO_H, LO_W, 0, wT_pw, b_pw, 256,
        512, 256, 1, 1, 1, 0, 0, LO_H, LO_W, 0, feat1, 256, 0, nullptr);

    // 7) directional convs -> dir buffer channel offsets 0/256/512
    conv_gemm<<<dim3(NT32, 4), 128, 0, stream>>>(
        feat1, 256, 0, LO_H, LO_W, 0, wT_d1, b_dir1, 256,
        256, 256, 1, 3, 1, 0, 1, LO_H, LO_W, 0, dir, 768, 0, nullptr);
    conv_gemm<<<dim3(NT32, 4), 128, 0, stream>>>(
        feat1, 256, 0, LO_H, LO_W, 0, wT_d2, b_dir2, 256,
        256, 256, 3, 1, 1, 1, 0, LO_H, LO_W, 0, dir, 768, 256, nullptr);
    conv_gemm<<<dim3(NT32, 4), 128, 0, stream>>>(
        feat1, 256, 0, LO_H, LO_W, 0, wT_d3, b_dir3, 256,
        256, 256, 3, 3, 2, 2, 2, LO_H, LO_W, 0, dir, 768, 512, nullptr);

    // 8) weight map: 768 -> 1, sigmoid
    wt_conv<<<ceildiv(NPIX, 256), 256, 0, stream>>>(dir, w_wt, b_wt, wgtb, LO_H, LO_W, NPIX);

    // 9) deformable conv + fused (* weight + f_low)
    deform_gemm<<<dim3(NT32, 4), 128, 0, stream>>>(
        cat, offsb, wgtb, wT_deform, modbuf, LO_H, LO_W);

    // 10) final 3x3 conv -> NCHW f32 output
    conv_gemm<<<dim3(NT32, 4), 128, 0, stream>>>(
        modbuf, 256, 0, LO_H, LO_W, 0, wT_final, b_final, 256,
        256, 256, 3, 3, 1, 1, 1, LO_H, LO_W, 2, nullptr, 0, 0, out);
}
